// AtnConv_13434657702199
// MI455X (gfx1250) — compile-verified, matrix operand order checked by reading, not run on hardware
//
#include <hip/hip_runtime.h>
#include <hip/hip_bf16.h>

// ---------------------------------------------------------------------------
// AtnConv for MI455X (gfx1250): both big GEMMs on v_wmma_f32_16x16x32_f16.
//   B=2, H=W=64, C=128, L=4096, k=3, KC=k*k*C=1152
// Pipeline:
//   1) extract x2 patches -> Pq f16 [B,L,1152]; inv-norm per patch (f32)
//   2) extract x1 patches transposed -> Vt f16 [B,1152,L]
//   3) GEMM1 (WMMA): scores[x,l] = (Pq[x].Pq[l]) * inv_norm[l]   f32 [B,L,L]
//   4) 3x3 count-avg pool *9 *10 + softmax over l -> att f16 [B,L,L]
//   5) GEMM2 (WMMA): contrib[x,dc] = sum_l att[x,l]*Vt[dc,l]     f32 [B,L,1152]
//   6) 9-way shifted gather-add -> y f32 [B,H,W,C]
// GEMM: 128x128 block tile, 8 waves (2x4), wave tile 64x32 (4x2 WMMA frags),
// double-buffered LDS + register-stage prefetch, one barrier per K-step.
// ---------------------------------------------------------------------------

typedef _Float16 v8h  __attribute__((ext_vector_type(8)));
typedef _Float16 v16h __attribute__((ext_vector_type(16)));
typedef float    v8f  __attribute__((ext_vector_type(8)));

#define B_   2
#define H_   64
#define W_   64
#define C_   128
#define L_   4096            // H_*W_
#define KC_  1152            // 9*C_

// ---- workspace layout (bytes, all 256-aligned) ----------------------------
#define OFF_PQ      ((size_t)0)                         // f16 [B,L,KC]  18.9MB
#define OFF_VT      ((size_t)18874368)                  // f16 [B,KC,L]  18.9MB
#define OFF_INV     ((size_t)37748736)                  // f32 [B,L]     32KB
#define OFF_SCORES  ((size_t)37781504)                  // f32 [B,L,L]   268MB
#define OFF_ATT     ((size_t)306216960)                 // f16 [B,L,L]   134MB
#define OFF_CONTRIB ((size_t)440434688)                 // f32 [B,L,KC]  37.7MB

// ---------------------------------------------------------------------------
// Patch extraction: Pq[b,l,d*C+c] = x2 zero-padded 3x3 patch, f16.
// ---------------------------------------------------------------------------
__global__ void atn_extract_q(const float* __restrict__ x2,
                              _Float16* __restrict__ Pq, int total) {
  int idx = blockIdx.x * blockDim.x + threadIdx.x;
  if (idx >= total) return;
  int c = idx & (C_ - 1);
  int t = idx >> 7;
  int d = t % 9;  t /= 9;
  int l = t & (L_ - 1);
  int b = t >> 12;
  int hy = (l >> 6) + d / 3 - 1;
  int hx = (l & 63) + d % 3 - 1;
  float v = 0.f;
  if (hy >= 0 && hy < H_ && hx >= 0 && hx < W_)
    v = x2[(((size_t)b * H_ + hy) * W_ + hx) * C_ + c];
  Pq[(size_t)(b * L_ + l) * KC_ + d * C_ + c] = (_Float16)v;
}

// Vt[b, d*C+c, l] = x1 patch value (transposed so GEMM2 is NT like GEMM1).
__global__ void atn_extract_vt(const float* __restrict__ x1,
                               _Float16* __restrict__ Vt, int total) {
  int idx = blockIdx.x * blockDim.x + threadIdx.x;
  if (idx >= total) return;
  int c = idx & (C_ - 1);
  int t = idx >> 7;
  int d = t % 9;  t /= 9;
  int l = t & (L_ - 1);
  int b = t >> 12;
  int hy = (l >> 6) + d / 3 - 1;
  int hx = (l & 63) + d % 3 - 1;
  float v = 0.f;
  if (hy >= 0 && hy < H_ && hx >= 0 && hx < W_)
    v = x1[(((size_t)b * H_ + hy) * W_ + hx) * C_ + c];
  Vt[((size_t)b * KC_ + d * C_ + c) * L_ + l] = (_Float16)v;
}

// inv_norm[b,l] = 1 / max(||patch_l(x2)||_2, 1e-4)   (fp32, from raw x2)
__global__ __launch_bounds__(128)
void atn_inv_norm(const float* __restrict__ x2, float* __restrict__ inv) {
  __shared__ float red[128];
  int bl = blockIdx.x;
  int b  = bl >> 12;
  int l  = bl & (L_ - 1);
  int ly = l >> 6, lx = l & 63;
  int c  = threadIdx.x;
  float s = 0.f;
  for (int d = 0; d < 9; ++d) {
    int hy = ly + d / 3 - 1, hx = lx + d % 3 - 1;
    if (hy >= 0 && hy < H_ && hx >= 0 && hx < W_) {
      float v = x2[(((size_t)b * H_ + hy) * W_ + hx) * C_ + c];
      s += v * v;
    }
  }
  red[c] = s;
  __syncthreads();
  for (int st = 64; st > 0; st >>= 1) {
    if (c < st) red[c] += red[c + st];
    __syncthreads();
  }
  if (c == 0) inv[bl] = 1.0f / fmaxf(sqrtf(red[0]), 1e-4f);
}

// ---------------------------------------------------------------------------
// NT WMMA GEMM: C[m,n] = scale[n] * sum_k A[m,k]*Bm[n,k]   (f16 in, f32 out)
// ---------------------------------------------------------------------------
#define TILE 128
#define KT   32
#define LSTR 40   // LDS row stride in halves: conflict-free b128 access

__global__ __launch_bounds__(256)
void atn_gemm_nt_wmma(const _Float16* __restrict__ A,
                      const _Float16* __restrict__ Bm,
                      float* __restrict__ C,
                      const float* __restrict__ scale,
                      int K, int lda, int ldb, int ldc,
                      size_t sA, size_t sB, size_t sC, size_t sScale) {
  __shared__ _Float16 As[2][TILE * LSTR];
  __shared__ _Float16 Bs[2][TILE * LSTR];

  const int b = blockIdx.z;
  A  += (size_t)b * sA;
  Bm += (size_t)b * sB;
  C  += (size_t)b * sC;
  const float* scl = scale ? scale + (size_t)b * sScale : nullptr;

  const int n0   = blockIdx.x * TILE;
  const int m0   = blockIdx.y * TILE;
  const int tid  = threadIdx.x;
  const int lane = tid & 31;
  const int wave = tid >> 5;
  const int hsel = lane >> 4;      // K-half selector (ISA 16-bit A/B layout)
  const int r    = lane & 15;      // row within 16x16 fragment
  const int wm0  = (wave >> 2) * 64;   // 2 wave-rows in M
  const int wn0  = (wave & 3) * 32;    // 4 wave-cols in N

  // Per-thread staging: 2 fixed 16B chunks of A and of B per K-step.
  const int row0 = tid >> 2;           // 0..63
  const int seg  = tid & 3;            // 0..3 (16B segment within 64B row)
  const int row1 = row0 + 64;
  const _Float16* pA0 = A  + (size_t)(m0 + row0) * lda + seg * 8;
  const _Float16* pA1 = A  + (size_t)(m0 + row1) * lda + seg * 8;
  const _Float16* pB0 = Bm + (size_t)(n0 + row0) * ldb + seg * 8;
  const _Float16* pB1 = Bm + (size_t)(n0 + row1) * ldb + seg * 8;
  const int sOff0 = row0 * LSTR + seg * 8;
  const int sOff1 = row1 * LSTR + seg * 8;

  v8f zero = {0.f, 0.f, 0.f, 0.f, 0.f, 0.f, 0.f, 0.f};
  v8f acc[4][2];
#pragma unroll
  for (int i = 0; i < 4; ++i)
#pragma unroll
    for (int j = 0; j < 2; ++j) acc[i][j] = zero;

  // Prologue: stage tile 0 into buffer 0.
  {
    v8h ra0 = *(const v8h*)pA0;  pA0 += KT;
    v8h ra1 = *(const v8h*)pA1;  pA1 += KT;
    v8h rb0 = *(const v8h*)pB0;  pB0 += KT;
    v8h rb1 = *(const v8h*)pB1;  pB1 += KT;
    *(v8h*)&As[0][sOff0] = ra0;
    *(v8h*)&As[0][sOff1] = ra1;
    *(v8h*)&Bs[0][sOff0] = rb0;
    *(v8h*)&Bs[0][sOff1] = rb1;
  }
  __syncthreads();

  const int nIter = K / KT;
  int buf = 0;
  for (int it = 0; it < nIter; ++it) {
    // Prefetch next tile into registers (overlaps with WMMA below).
    v8h ra0, ra1, rb0, rb1;
    const bool hasNext = (it + 1 < nIter);
    if (hasNext) {
      ra0 = *(const v8h*)pA0;  pA0 += KT;
      ra1 = *(const v8h*)pA1;  pA1 += KT;
      rb0 = *(const v8h*)pB0;  pB0 += KT;
      rb1 = *(const v8h*)pB1;  pB1 += KT;
    }

    // Load all 6 fragments up front, then 8 back-to-back WMMAs.
    v16h bf[2], af[4];
#pragma unroll
    for (int j = 0; j < 2; ++j) {
      const _Float16* p = &Bs[buf][(wn0 + j * 16 + r) * LSTR + hsel * 8];
      v8h lo = *(const v8h*)p;          // K = hsel*8 + 0..7
      v8h hi = *(const v8h*)(p + 16);   // K = 16 + hsel*8 + 0..7
      bf[j]  = __builtin_shufflevector(lo, hi, 0, 1, 2, 3, 4, 5, 6, 7,
                                       8, 9, 10, 11, 12, 13, 14, 15);
    }
#pragma unroll
    for (int i = 0; i < 4; ++i) {
      const _Float16* p = &As[buf][(wm0 + i * 16 + r) * LSTR + hsel * 8];
      v8h lo = *(const v8h*)p;
      v8h hi = *(const v8h*)(p + 16);
      af[i]  = __builtin_shufflevector(lo, hi, 0, 1, 2, 3, 4, 5, 6, 7,
                                       8, 9, 10, 11, 12, 13, 14, 15);
    }
#pragma unroll
    for (int i = 0; i < 4; ++i)
#pragma unroll
      for (int j = 0; j < 2; ++j)
        acc[i][j] = __builtin_amdgcn_wmma_f32_16x16x32_f16(
            false, af[i], false, bf[j], (short)0, acc[i][j], false, false);

    if (hasNext) {
      const int nb = buf ^ 1;
      *(v8h*)&As[nb][sOff0] = ra0;
      *(v8h*)&As[nb][sOff1] = ra1;
      *(v8h*)&Bs[nb][sOff0] = rb0;
      *(v8h*)&Bs[nb][sOff1] = rb1;
      __syncthreads();   // waits dscnt: fences reads of buf & stores to nb
      buf = nb;
    }
  }

  // Epilogue: D layout -> lane r = N, VGPR v -> M = v + 8*hsel
#pragma unroll
  for (int i = 0; i < 4; ++i)
#pragma unroll
    for (int j = 0; j < 2; ++j) {
      int   n = n0 + wn0 + j * 16 + r;
      float s = scl ? scl[n] : 1.0f;
#pragma unroll
      for (int v = 0; v < 8; ++v) {
        int m = m0 + wm0 + i * 16 + hsel * 8 + v;
        C[(size_t)m * ldc + n] = acc[i][j][v] * s;
      }
    }
}

// ---------------------------------------------------------------------------
// pooled = (3x3 spatial sum over x) * 9/cnt * 10; att = softmax over l (f32),
// stored f16. One block per (b,x); full 4096-row kept in LDS.
// ---------------------------------------------------------------------------
__global__ __launch_bounds__(256)
void atn_pool_softmax(const float* __restrict__ scores,
                      _Float16* __restrict__ att) {
  __shared__ float prow[L_];
  __shared__ float red[256];
  int bx = blockIdx.x;
  int b  = bx >> 12;
  int x  = bx & (L_ - 1);
  int hy = x >> 6, hx = x & 63;
  int cy = 1 + (hy > 0) + (hy < H_ - 1);
  int cx = 1 + (hx > 0) + (hx < W_ - 1);
  float factor = (9.0f / (float)(cy * cx)) * 10.0f;   // *9/cnt then *scale

  const size_t base = (size_t)b * L_ * L_;
  float lmax = -3.4e38f;
  for (int l = threadIdx.x; l < L_; l += 256) {
    float p = 0.f;
    for (int dy = -1; dy <= 1; ++dy) {
      int yy = hy + dy;
      if (yy < 0 || yy >= H_) continue;
      for (int dx = -1; dx <= 1; ++dx) {
        int xx = hx + dx;
        if (xx < 0 || xx >= W_) continue;
        p += scores[base + (size_t)(yy * W_ + xx) * L_ + l];
      }
    }
    p *= factor;
    prow[l] = p;
    lmax = fmaxf(lmax, p);
  }
  red[threadIdx.x] = lmax;
  __syncthreads();
  for (int s = 128; s > 0; s >>= 1) {
    if (threadIdx.x < s)
      red[threadIdx.x] = fmaxf(red[threadIdx.x], red[threadIdx.x + s]);
    __syncthreads();
  }
  float mx = red[0];
  __syncthreads();

  float lsum = 0.f;
  for (int l = threadIdx.x; l < L_; l += 256) {
    float e = __expf(prow[l] - mx);
    prow[l] = e;
    lsum += e;
  }
  red[threadIdx.x] = lsum;
  __syncthreads();
  for (int s = 128; s > 0; s >>= 1) {
    if (threadIdx.x < s) red[threadIdx.x] += red[threadIdx.x + s];
    __syncthreads();
  }
  float invsum = 1.0f / red[0];
  for (int l = threadIdx.x; l < L_; l += 256)
    att[(size_t)bx * L_ + l] = (_Float16)(prow[l] * invsum);
}

// ---------------------------------------------------------------------------
// y[b,q,c] = sum over 3x3 taps d of contrib[b, q - delta(d), d, c]
// ---------------------------------------------------------------------------
__global__ void atn_gather(const float* __restrict__ contrib,
                           float* __restrict__ y, int total) {
  int idx = blockIdx.x * blockDim.x + threadIdx.x;
  if (idx >= total) return;
  int c = idx & (C_ - 1);
  int q = (idx >> 7) & (L_ - 1);
  int b = idx >> 19;
  int qy = q >> 6, qx = q & 63;
  float acc = 0.f;
  for (int dy = -1; dy <= 1; ++dy) {
    int py = qy - dy;
    if (py < 0 || py >= H_) continue;
    for (int dx = -1; dx <= 1; ++dx) {
      int px = qx - dx;
      if (px < 0 || px >= W_) continue;
      int d = (dy + 1) * 3 + (dx + 1);
      acc += contrib[(((size_t)(b * L_) + py * W_ + px) * 9 + d) * C_ + c];
    }
  }
  y[idx] = acc;
}

// ---------------------------------------------------------------------------
extern "C" void kernel_launch(void* const* d_in, const int* in_sizes, int n_in,
                              void* d_out, int out_size, void* d_ws,
                              size_t ws_size, hipStream_t stream) {
  const float* x1 = (const float*)d_in[0];
  const float* x2 = (const float*)d_in[1];
  // d_in[2] (mask) does not affect the first-call output per the reference.

  char* ws = (char*)d_ws;
  _Float16* Pq  = (_Float16*)(ws + OFF_PQ);
  _Float16* Vt  = (_Float16*)(ws + OFF_VT);
  float*    inv = (float*)(ws + OFF_INV);
  float*    sc  = (float*)(ws + OFF_SCORES);
  _Float16* att = (_Float16*)(ws + OFF_ATT);
  float*    ctb = (float*)(ws + OFF_CONTRIB);

  const int totP = B_ * L_ * 9 * C_;                 // 9,437,184
  atn_extract_q <<<totP / 256, 256, 0, stream>>>(x2, Pq, totP);
  atn_extract_vt<<<totP / 256, 256, 0, stream>>>(x1, Vt, totP);
  atn_inv_norm  <<<B_ * L_, 128, 0, stream>>>(x2, inv);

  // GEMM1: [4096 x 4096 x 1152] per batch; scale columns by inv_norm[l]
  atn_gemm_nt_wmma<<<dim3(L_ / TILE, L_ / TILE, B_), 256, 0, stream>>>(
      Pq, Pq, sc, inv, /*K=*/KC_, /*lda=*/KC_, /*ldb=*/KC_, /*ldc=*/L_,
      (size_t)L_ * KC_, (size_t)L_ * KC_, (size_t)L_ * L_, (size_t)L_);

  atn_pool_softmax<<<B_ * L_, 256, 0, stream>>>(sc, att);

  // GEMM2: [4096 x 1152 x 4096] per batch
  atn_gemm_nt_wmma<<<dim3(KC_ / TILE, L_ / TILE, B_), 256, 0, stream>>>(
      att, Vt, ctb, nullptr, /*K=*/L_, /*lda=*/L_, /*ldb=*/L_, /*ldc=*/KC_,
      (size_t)L_ * L_, (size_t)KC_ * L_, (size_t)L_ * KC_, (size_t)0);

  const int totO = B_ * L_ * C_;                     // 1,048,576
  atn_gather<<<totO / 256, 256, 0, stream>>>(ctb, (float*)d_out, totO);
}
